// SVDHead_76871324664470
// MI455X (gfx1250) — matrix operand order repaired and checked by reading, not compiled, not run on hardware
//
#include <hip/hip_runtime.h>

// Problem sizes (fixed by the reference).
#define B_ROWS 32768
#define D_IN   1024
#define H_MID  2048
#define LN_EPS 1e-5f

typedef __attribute__((ext_vector_type(16))) __bf16          v16bf;
typedef __attribute__((ext_vector_type(8)))  float           v8f;
typedef __attribute__((ext_vector_type(8)))  unsigned short  u16x8;
typedef __attribute__((ext_vector_type(16))) unsigned short  u16x16;

union BF16Frag { u16x16 u; v16bf b; };

__device__ __forceinline__ unsigned short f2bf(float f) {
  unsigned int u = __float_as_uint(f);
  u += 0x7fffu + ((u >> 16) & 1u);   // round-to-nearest-even
  return (unsigned short)(u >> 16);
}
__device__ __forceinline__ float bf2f(unsigned short h) {
  return __uint_as_float(((unsigned int)h) << 16);
}

// Two 16-byte (b128) loads -> one 32-byte bf16 fragment.
__device__ __forceinline__ u16x16 ld_frag(const unsigned short* __restrict__ p0,
                                          const unsigned short* __restrict__ p1) {
  u16x8 lo = *(const u16x8*)p0;
  u16x8 hi = *(const u16x8*)p1;
  u16x16 r;
#pragma unroll
  for (int i = 0; i < 8; ++i) { r[i] = lo[i]; r[i + 8] = hi[i]; }
  return r;
}

// ---------------------------------------------------------------------------
// Prep kernels
// ---------------------------------------------------------------------------
__global__ void f32_to_bf16_kernel(const float* __restrict__ in,
                                   unsigned short* __restrict__ out, int n4) {
  int i = blockIdx.x * blockDim.x + threadIdx.x;
  int stride = gridDim.x * blockDim.x;
  for (; i < n4; i += stride) {
    float4 f = ((const float4*)in)[i];
    ushort4 o;
    o.x = f2bf(f.x); o.y = f2bf(f.y); o.z = f2bf(f.z); o.w = f2bf(f.w);
    ((ushort4*)out)[i] = o;
  }
}

// w1t[n][k] = bf16(W1[k][n]) : B-operand friendly (per-column contiguous K).
__global__ void pack_w1t_kernel(const float* __restrict__ W1,
                                unsigned short* __restrict__ w1t) {
  int idx = blockIdx.x * 256 + threadIdx.x;     // over H_MID * D_IN
  int n = idx >> 10;                            // / D_IN
  int k = idx & (D_IN - 1);
  w1t[idx] = f2bf(W1[(size_t)k * H_MID + n]);
}

// gw2t[n][k] = bf16(g1[k] * W2[k][n]) for n<9, else 0. Padded to 16 columns.
__global__ void pack_gw2_kernel(const float* __restrict__ W2,
                                const float* __restrict__ g1,
                                unsigned short* __restrict__ gw2t) {
  int idx = blockIdx.x * 256 + threadIdx.x;     // over 16 * H_MID
  int n = idx >> 11;                            // / H_MID
  int k = idx & (H_MID - 1);
  float v = (n < 9) ? g1[k] * W2[(size_t)k * 9 + n] : 0.0f;
  gw2t[idx] = f2bf(v);
}

// consts[j]    = C1_j = sum_k g1[k]  * W2[k][j]
// consts[16+j] = C2_j = sum_k be1[k] * W2[k][j]
__global__ void consts_kernel(const float* __restrict__ W2,
                              const float* __restrict__ g1,
                              const float* __restrict__ be1,
                              float* __restrict__ consts) {
  __shared__ float s1[256], s2[256];
  const int j = blockIdx.x;   // 0..8
  float a = 0.f, b = 0.f;
  for (int k = threadIdx.x; k < H_MID; k += 256) {
    float w = W2[(size_t)k * 9 + j];
    a = fmaf(g1[k], w, a);
    b = fmaf(be1[k], w, b);
  }
  s1[threadIdx.x] = a; s2[threadIdx.x] = b;
  __syncthreads();
  for (int off = 128; off > 0; off >>= 1) {
    if (threadIdx.x < off) {
      s1[threadIdx.x] += s1[threadIdx.x + off];
      s2[threadIdx.x] += s2[threadIdx.x + off];
    }
    __syncthreads();
  }
  if (threadIdx.x == 0) { consts[j] = s1[0]; consts[16 + j] = s2[0]; }
}

// ---------------------------------------------------------------------------
// GEMM1: h = relu(x @ W1 + b1), stored as bf16 [B, H].
// Block = 256 threads (8 waves), block tile 128x256, wave tile 64x64.
// 16 v_wmma_f32_16x16x32_bf16 per 32-wide K step.
// ---------------------------------------------------------------------------
__global__ __launch_bounds__(256) void gemm1_relu_kernel(
    const unsigned short* __restrict__ xbf,
    const unsigned short* __restrict__ w1t,
    const float* __restrict__ b1,
    unsigned short* __restrict__ abf) {
  const int lane = threadIdx.x & 31;
  const int wave = threadIdx.x >> 5;
  const int l15  = lane & 15;
  const int lh   = lane >> 4;                 // which half of the wave
  const int rowBase = blockIdx.x * 128 + (wave >> 2) * 64;
  const int colBase = blockIdx.y * 256 + (wave & 3) * 64;

  v8f acc[4][4];
#pragma unroll
  for (int mi = 0; mi < 4; ++mi)
#pragma unroll
    for (int ni = 0; ni < 4; ++ni) acc[mi][ni] = (v8f)0.0f;

  // A fragment: lane l (l<16) carries row l; halves split K 0..7 / 8..15 (+16).
  const unsigned short* aBase = xbf + (size_t)(rowBase + l15) * D_IN + lh * 8;
  // B fragment: lane carries column; halves split K 0..7 / 8..15 (+16).
  const unsigned short* bBase = w1t + (size_t)(colBase + l15) * D_IN + lh * 8;

  for (int k = 0; k < D_IN; k += 32) {
    BF16Frag a[4], b[4];
#pragma unroll
    for (int mi = 0; mi < 4; ++mi) {
      const unsigned short* p = aBase + (size_t)mi * 16 * D_IN + k;
      a[mi].u = ld_frag(p, p + 16);
    }
#pragma unroll
    for (int ni = 0; ni < 4; ++ni) {
      const unsigned short* p = bBase + (size_t)ni * 16 * D_IN + k;
      b[ni].u = ld_frag(p, p + 16);
    }
#pragma unroll
    for (int mi = 0; mi < 4; ++mi)
#pragma unroll
      for (int ni = 0; ni < 4; ++ni)
        acc[mi][ni] = __builtin_amdgcn_wmma_f32_16x16x32_bf16(
            false, a[mi].b, false, b[ni].b, (short)0, acc[mi][ni], false, false);
  }

  // Epilogue: bias + ReLU + bf16 store. D layout: lanes 0-15 -> N=lane, VGPR v -> M=v;
  // lanes 16-31 -> M=v+8.
#pragma unroll
  for (int ni = 0; ni < 4; ++ni) {
    const int col = colBase + ni * 16 + l15;
    const float bias = b1[col];
#pragma unroll
    for (int mi = 0; mi < 4; ++mi) {
#pragma unroll
      for (int v = 0; v < 8; ++v) {
        const int row = rowBase + mi * 16 + v + lh * 8;
        float val = fmaxf(acc[mi][ni][v] + bias, 0.0f);
        abf[(size_t)row * H_MID + col] = f2bf(val);
      }
    }
  }
}

// ---------------------------------------------------------------------------
// 3x3 SVD -> SO(3) projection (per lane, fully unrolled, no scratch).
// ---------------------------------------------------------------------------
template <int P, int Q>
__device__ __forceinline__ void jrot(float a[3][3], float v[3][3]) {
  float apq = a[P][Q];
  if (fabsf(apq) < 1e-12f) return;
  float theta = (a[Q][Q] - a[P][P]) / (2.0f * apq);
  float t = copysignf(1.0f, theta) / (fabsf(theta) + sqrtf(theta * theta + 1.0f));
  float c = rsqrtf(t * t + 1.0f);
  float s = t * c;
#pragma unroll
  for (int k = 0; k < 3; ++k) {
    float akp = a[k][P], akq = a[k][Q];
    a[k][P] = c * akp - s * akq;
    a[k][Q] = s * akp + c * akq;
  }
#pragma unroll
  for (int k = 0; k < 3; ++k) {
    float apk = a[P][k], aqk = a[Q][k];
    a[P][k] = c * apk - s * aqk;
    a[Q][k] = s * apk + c * aqk;
  }
#pragma unroll
  for (int k = 0; k < 3; ++k) {
    float vkp = v[k][P], vkq = v[k][Q];
    v[k][P] = c * vkp - s * vkq;
    v[k][Q] = s * vkp + c * vkq;
  }
}

template <int I, int J>
__device__ __forceinline__ void swapcol(float v[3][3]) {
#pragma unroll
  for (int k = 0; k < 3; ++k) { float t = v[k][I]; v[k][I] = v[k][J]; v[k][J] = t; }
}

__device__ __forceinline__ void so3_project(const float R[9], float Pm[9]) {
  float a[3][3], v[3][3];
#pragma unroll
  for (int i = 0; i < 3; ++i)
#pragma unroll
    for (int j = 0; j < 3; ++j) {
      float s = 0.f;
#pragma unroll
      for (int k = 0; k < 3; ++k) s = fmaf(R[k * 3 + i], R[k * 3 + j], s);  // R^T R
      a[i][j] = s;
      v[i][j] = (i == j) ? 1.f : 0.f;
    }
#pragma unroll
  for (int sweep = 0; sweep < 5; ++sweep) {
    jrot<0, 1>(a, v); jrot<0, 2>(a, v); jrot<1, 2>(a, v);
  }
  float l0 = a[0][0], l1 = a[1][1], l2 = a[2][2];
  if (l0 < l1) { float t = l0; l0 = l1; l1 = t; swapcol<0, 1>(v); }
  if (l0 < l2) { float t = l0; l0 = l2; l2 = t; swapcol<0, 2>(v); }
  if (l1 < l2) { float t = l1; l1 = l2; l2 = t; swapcol<1, 2>(v); }

  float u1[3], u2[3], u3[3];
#pragma unroll
  for (int i = 0; i < 3; ++i)
    u1[i] = R[i * 3 + 0] * v[0][0] + R[i * 3 + 1] * v[1][0] + R[i * 3 + 2] * v[2][0];
  float n1 = rsqrtf(fmaxf(u1[0] * u1[0] + u1[1] * u1[1] + u1[2] * u1[2], 1e-20f));
#pragma unroll
  for (int i = 0; i < 3; ++i) u1[i] *= n1;
#pragma unroll
  for (int i = 0; i < 3; ++i)
    u2[i] = R[i * 3 + 0] * v[0][1] + R[i * 3 + 1] * v[1][1] + R[i * 3 + 2] * v[2][1];
  float d12 = u1[0] * u2[0] + u1[1] * u2[1] + u1[2] * u2[2];
#pragma unroll
  for (int i = 0; i < 3; ++i) u2[i] -= d12 * u1[i];
  float n2 = rsqrtf(fmaxf(u2[0] * u2[0] + u2[1] * u2[1] + u2[2] * u2[2], 1e-20f));
#pragma unroll
  for (int i = 0; i < 3; ++i) u2[i] *= n2;
  u3[0] = u1[1] * u2[2] - u1[2] * u2[1];
  u3[1] = u1[2] * u2[0] - u1[0] * u2[2];
  u3[2] = u1[0] * u2[1] - u1[1] * u2[0];
  // d = det(V) (det(U)=1 by construction) == det(U V^T)
  float cx = v[1][1] * v[2][2] - v[2][1] * v[1][2];
  float cy = v[2][1] * v[0][2] - v[0][1] * v[2][2];
  float cz = v[0][1] * v[1][2] - v[1][1] * v[0][2];
  float dv = v[0][0] * cx + v[1][0] * cy + v[2][0] * cz;
  float d = (dv >= 0.f) ? 1.f : -1.f;
#pragma unroll
  for (int i = 0; i < 3; ++i)
#pragma unroll
    for (int j = 0; j < 3; ++j)
      Pm[i * 3 + j] = u1[i] * v[j][0] + u2[i] * v[j][1] + d * u3[i] * v[j][2];
}

// ---------------------------------------------------------------------------
// Head: per 16-row wave tile, S = a @ (g1*W2) via WMMA + fused row stats,
// then LN algebra, ReLU, LN(9), SVD projection, store [B,3,3].
// ---------------------------------------------------------------------------
__global__ __launch_bounds__(256) void head_kernel(
    const unsigned short* __restrict__ abf,
    const unsigned short* __restrict__ gw2t,
    const float* __restrict__ consts,
    const float* __restrict__ b2,
    const float* __restrict__ g2,
    const float* __restrict__ be2,
    float* __restrict__ out) {
  __shared__ float smem[8][16][17];
  const int lane = threadIdx.x & 31;
  const int wave = threadIdx.x >> 5;
  const int l15  = lane & 15;
  const int lh   = lane >> 4;
  const int rowBase = blockIdx.x * 128 + wave * 16;

  v8f acc = (v8f)0.0f;
  float rs = 0.f, rq = 0.f;
  const unsigned short* aBase = abf + (size_t)(rowBase + l15) * H_MID + lh * 8;
  const unsigned short* bBase = gw2t + (size_t)l15 * H_MID + lh * 8;

  for (int k = 0; k < H_MID; k += 32) {
    BF16Frag a, b;
    { const unsigned short* p = aBase + k; a.u = ld_frag(p, p + 16); }
    { const unsigned short* p = bBase + k; b.u = ld_frag(p, p + 16); }
    // Row sum / sumsq straight off the A fragment (halves of each row combine below).
#pragma unroll
    for (int i = 0; i < 16; ++i) {
      float f = bf2f(a.u[i]);
      rs += f;
      rq = fmaf(f, f, rq);
    }
    acc = __builtin_amdgcn_wmma_f32_16x16x32_bf16(
        false, a.b, false, b.b, (short)0, acc, false, false);
  }
  rs += __shfl_xor(rs, 16);
  rq += __shfl_xor(rq, 16);
  const float m   = rs * (1.0f / (float)H_MID);
  const float var = rq * (1.0f / (float)H_MID) - m * m;
  const float inv = rsqrtf(var + LN_EPS);

  // D-layout -> per-row handoff through LDS.
#pragma unroll
  for (int v = 0; v < 8; ++v) smem[wave][v + lh * 8][l15] = acc[v];
  __syncthreads();

  if (lane < 16) {
    const int row = rowBase + lane;
    float t[9];
    float s9 = 0.f, q9 = 0.f;
#pragma unroll
    for (int j = 0; j < 9; ++j) {
      float S = smem[wave][lane][j];
      float r = inv * S - inv * m * consts[j] + consts[16 + j] + b2[j];
      r = fmaxf(r, 0.0f);
      t[j] = r;
      s9 += r;
      q9 = fmaf(r, r, q9);
    }
    const float m9 = s9 * (1.0f / 9.0f);
    const float v9 = q9 * (1.0f / 9.0f) - m9 * m9;
    const float i9 = rsqrtf(v9 + LN_EPS);
    float Rm[9], Pm[9];
#pragma unroll
    for (int j = 0; j < 9; ++j) Rm[j] = (t[j] - m9) * i9 * g2[j] + be2[j];
    so3_project(Rm, Pm);
#pragma unroll
    for (int j = 0; j < 9; ++j) out[(size_t)row * 9 + j] = Pm[j];
  }
}

// ---------------------------------------------------------------------------
// Launch
// ---------------------------------------------------------------------------
extern "C" void kernel_launch(void* const* d_in, const int* in_sizes, int n_in,
                              void* d_out, int out_size, void* d_ws, size_t ws_size,
                              hipStream_t stream) {
  (void)in_sizes; (void)n_in; (void)out_size; (void)ws_size;
  const float* x   = (const float*)d_in[0];
  const float* W1  = (const float*)d_in[1];
  const float* b1  = (const float*)d_in[2];
  const float* g1  = (const float*)d_in[3];
  const float* be1 = (const float*)d_in[4];
  const float* W2  = (const float*)d_in[5];
  const float* b2  = (const float*)d_in[6];
  const float* g2  = (const float*)d_in[7];
  const float* be2 = (const float*)d_in[8];
  float* out = (float*)d_out;

  char* ws = (char*)d_ws;
  size_t off = 0;
  unsigned short* xbf  = (unsigned short*)(ws + off); off += (size_t)B_ROWS * D_IN * 2;   // 64 MB
  unsigned short* w1t  = (unsigned short*)(ws + off); off += (size_t)D_IN * H_MID * 2;    // 4 MB
  unsigned short* gw2t = (unsigned short*)(ws + off); off += (size_t)16 * H_MID * 2;      // 64 KB
  float*          cst  = (float*)(ws + off);          off += 256;
  unsigned short* abf  = (unsigned short*)(ws + off); off += (size_t)B_ROWS * H_MID * 2;  // 128 MB

  f32_to_bf16_kernel<<<4096, 256, 0, stream>>>(x, xbf, (B_ROWS * D_IN) / 4);
  pack_w1t_kernel<<<(D_IN * H_MID) / 256, 256, 0, stream>>>(W1, w1t);
  pack_gw2_kernel<<<(16 * H_MID) / 256, 256, 0, stream>>>(W2, g1, gw2t);
  consts_kernel<<<9, 256, 0, stream>>>(W2, g1, be1, cst);
  gemm1_relu_kernel<<<dim3(B_ROWS / 128, H_MID / 256), 256, 0, stream>>>(xbf, w1t, b1, abf);
  head_kernel<<<B_ROWS / 128, 256, 0, stream>>>(abf, gw2t, cst, b2, g2, be2, out);
}
